// MultiHeadAttention_65180423684703
// MI455X (gfx1250) — compile-verified, matrix operand order checked by reading, not run on hardware
//
#include <hip/hip_runtime.h>
#include <hip/hip_bf16.h>

// MI455X / gfx1250, wave32. Multi-head attention scores + softmax.
// B=2, S=2048, D=768, H=12, DK=64.
//
// Pipeline:
//   proj_kernel  x2 : q = query@Wq.T + bq, k = key@Wk.T + bk   (WMMA f16->f32)
//                     stored f16 in head-major [B,H,S,DK] in d_ws (12 MB, L2-resident)
//   attn_kernel     : per (b,h,16-query-tile): scores via WMMA over DK=64,
//                     mask penalties, softmax in LDS, single coalesced HBM write.
//
// Both kernels use dual independent accumulators per wave so consecutive
// v_wmma ops never form a D->C RAW chain (avoids the 4-5 cycle hazard NOPs).

typedef __attribute__((ext_vector_type(16))) _Float16 v16h;
typedef __attribute__((ext_vector_type(8)))  float    v8f;

#define NB    2
#define NS    2048
#define ND    768
#define NH    12
#define NDK   64
#define SENT  (-10000.0f)
#define PEN   (1.0e9f)

// --- WMMA 16x16x32 f16 fragment loaders -------------------------------------
// A (16xK) and B (KxN) share the same per-lane layout on CDNA5:
//   lane&15 selects the row (A) / column (B); lane>>4 selects the K half.
//   VGPR pair i (i=0..7) holds K = (i>>2)*16 + (lane>>4)*8 + (i&3)*2 (+1).

__device__ __forceinline__ v16h load_frag_f32(const float* __restrict__ src,
                                              int row_stride, int rbase,
                                              int kbase, int lane) {
  const int g = (lane >> 4) & 1;
  const int r = lane & 15;
  const float* p = src + (size_t)(rbase + r) * row_stride + kbase;
  v16h f;
#pragma unroll
  for (int i = 0; i < 8; ++i) {
    const int k = ((i >> 2) << 4) + (g << 3) + ((i & 3) << 1);
    const float2 d = *(const float2*)(p + k);   // merged into global_load_b128
    f[2 * i]     = (_Float16)d.x;
    f[2 * i + 1] = (_Float16)d.y;
  }
  return f;
}

__device__ __forceinline__ v16h load_frag_f16(const _Float16* __restrict__ src,
                                              int row_stride, int rbase,
                                              int kbase, int lane) {
  const int g = (lane >> 4) & 1;
  const int r = lane & 15;
  const _Float16* p = src + (size_t)(rbase + r) * row_stride + kbase;
  v16h f;
#pragma unroll
  for (int i = 0; i < 8; ++i) {
    const int k = ((i >> 2) << 4) + (g << 3) + ((i & 3) << 1);   // even -> 4B aligned
    const unsigned u = *(const unsigned*)(p + k);
    f[2 * i]     = ((const _Float16*)&u)[0];
    f[2 * i + 1] = ((const _Float16*)&u)[1];
  }
  return f;
}

__device__ __forceinline__ v8f wmma_f16(v16h a, v16h b, v8f c) {
  return __builtin_amdgcn_wmma_f32_16x16x32_f16(
      /*neg_a=*/false, a, /*neg_b=*/false, b,
      /*c_mod=*/(short)0, c, /*reuse_a=*/false, /*reuse_b=*/false);
}

// --- Projection: out[m,n] = sum_k in[m,k]*W[n,k] + bias[n] ------------------
// One wave per 16x32 output slab (two N-tiles, dual accumulators sharing A).
// M = B*S = 4096 (256 tiles), N = 768 (24 tile-pairs). 6144 waves.
__global__ __launch_bounds__(128) void proj_kernel(
    const float* __restrict__ in, const float* __restrict__ W,
    const float* __restrict__ bias, _Float16* __restrict__ outh) {
  const int lane   = threadIdx.x & 31;
  const int waveId = (blockIdx.x << 2) + (threadIdx.x >> 5);
  const int mtile  = waveId & 255;
  const int npair  = waveId >> 8;         // 0..23
  const int mbase  = mtile << 4;
  const int nbase0 = npair << 5;
  const int nbase1 = nbase0 + 16;

  v8f c0 = {}, c1 = {};
#pragma unroll 4
  for (int ks = 0; ks < ND; ks += 32) {
    const v16h a   = load_frag_f32(in, ND, mbase,  ks, lane);  // shared activations
    const v16h bf0 = load_frag_f32(W,  ND, nbase0, ks, lane);  // weights tile 0
    const v16h bf1 = load_frag_f32(W,  ND, nbase1, ks, lane);  // weights tile 1
    c0 = wmma_f16(a, bf0, c0);   // independent chains: no WMMA->WMMA RAW
    c1 = wmma_f16(a, bf1, c1);
  }

  // C layout: lane&15 = N, vgpr r -> M = r + 8*(lane>>4).
  const int col = lane & 15;
  const int g   = (lane >> 4) & 1;
#pragma unroll
  for (int t = 0; t < 2; ++t) {
    const int n  = (t == 0 ? nbase0 : nbase1) + col;
    const float bv = bias[n];
    const int h  = n >> 6;      // n / DK
    const int dk = n & 63;      // n % DK
    const v8f& c = (t == 0) ? c0 : c1;
#pragma unroll
    for (int r = 0; r < 8; ++r) {
      const int m = mbase + r + (g << 3);
      const int bb = m >> 11;           // m / S
      const int s  = m & (NS - 1);      // m % S
      outh[(((size_t)(bb * NH + h) * NS) + s) * NDK + dk] = (_Float16)(c[r] + bv);
    }
  }
}

// --- Scores + softmax --------------------------------------------------------
// One 128-thread workgroup per (b,h, 16-query tile). Scores staged in LDS
// (16 rows x 2048 keys, padded), softmax fully on-chip, single HBM write.
#define SROW 2064   // 2048 + 16 dword padding to break bank-conflict alignment

__global__ __launch_bounds__(128) void attn_kernel(
    const _Float16* __restrict__ qh, const _Float16* __restrict__ kh,
    const float* __restrict__ mask, float* __restrict__ out) {
  __shared__ float s_scores[16 * SROW];   // 132096 B (<= 320KB WG limit on CDNA5)
  __shared__ float s_inv[16];
  __shared__ float s_penq[16];

  const int tid  = threadIdx.x;
  const int wave = tid >> 5;
  const int lane = tid & 31;
  const int bh    = blockIdx.x >> 7;      // 128 query tiles per (b,h)
  const int qtile = blockIdx.x & 127;
  const int b     = bh / NH;
  const int qbase = qtile << 4;

  if (tid < 16) {
    const float mv = mask[b * NS + qbase + tid];
    s_penq[tid] = (mv == SENT) ? PEN : mv;
  }

  const _Float16* qptr = qh + (size_t)bh * NS * NDK;
  const _Float16* kptr = kh + (size_t)bh * NS * NDK;

  // Q fragment: A matrix 16x64 (two K=32 halves), reused for all key tiles.
  const v16h a0 = load_frag_f16(qptr, NDK, qbase, 0,  lane);
  const v16h a1 = load_frag_f16(qptr, NDK, qbase, 32, lane);

  __syncthreads();   // s_penq visible

  const int col = lane & 15;
  const int g   = (lane >> 4) & 1;

  // Each wave handles key-tile pairs {8i+2w, 8i+2w+1}: disjoint cover of 128
  // tiles across 4 waves, dual accumulators to break the WMMA hazard chain.
  for (int kt = wave << 1; kt < NS / 16; kt += 8) {
    const int kb0 = kt << 4;
    const int kb1 = kb0 + 16;
    const v16h b00 = load_frag_f16(kptr, NDK, kb0, 0,  lane);  // B[d][kj]=k[kj][d]
    const v16h b01 = load_frag_f16(kptr, NDK, kb0, 32, lane);
    const v16h b10 = load_frag_f16(kptr, NDK, kb1, 0,  lane);
    const v16h b11 = load_frag_f16(kptr, NDK, kb1, 32, lane);
    v8f c0 = {}, c1 = {};
    c0 = wmma_f16(a0, b00, c0);   // alternate chains -> hazard hidden
    c1 = wmma_f16(a0, b10, c1);
    c0 = wmma_f16(a1, b01, c0);
    c1 = wmma_f16(a1, b11, c1);

    const float mv0   = mask[b * NS + kb0 + col];
    const float mv1   = mask[b * NS + kb1 + col];
    const float penk0 = (mv0 == SENT) ? PEN : mv0;
    const float penk1 = (mv1 == SENT) ? PEN : mv1;
#pragma unroll
    for (int r = 0; r < 8; ++r) {
      const int q = r + (g << 3);
      s_scores[q * SROW + kb0 + col] = c0[r] * 0.125f - s_penq[q] - penk0;
      s_scores[q * SROW + kb1 + col] = c1[r] * 0.125f - s_penq[q] - penk1;
    }
  }
  __syncthreads();

  // Softmax: 8 lanes per row (rows 0..15), shuffle reductions within the wave.
  const int row = tid >> 3;
  const int j   = tid & 7;
  float* rowp = s_scores + row * SROW;

  float mx = -3.4e38f;
#pragma unroll 8
  for (int i = 0; i < NS / 8; ++i) mx = fmaxf(mx, rowp[j + (i << 3)]);
  mx = fmaxf(mx, __shfl_xor(mx, 1));
  mx = fmaxf(mx, __shfl_xor(mx, 2));
  mx = fmaxf(mx, __shfl_xor(mx, 4));

  float sum = 0.0f;
#pragma unroll 8
  for (int i = 0; i < NS / 8; ++i) {
    const float x = __expf(rowp[j + (i << 3)] - mx);
    rowp[j + (i << 3)] = x;
    sum += x;
  }
  sum += __shfl_xor(sum, 1);
  sum += __shfl_xor(sum, 2);
  sum += __shfl_xor(sum, 4);
  if (j == 0) s_inv[row] = 1.0f / sum;
  __syncthreads();

  // Single coalesced store of the normalized tile: this is the HBM roofline term.
  float* obase = out + ((size_t)bh * NS + qbase) * NS;
  for (int idx = tid; idx < 16 * NS; idx += 128) {
    const int q   = idx >> 11;
    const int key = idx & (NS - 1);
    obase[(size_t)q * NS + key] = s_scores[q * SROW + key] * s_inv[q];
  }
}

extern "C" void kernel_launch(void* const* d_in, const int* in_sizes, int n_in,
                              void* d_out, int out_size, void* d_ws, size_t ws_size,
                              hipStream_t stream) {
  (void)in_sizes; (void)n_in; (void)out_size; (void)ws_size;
  const float* query = (const float*)d_in[0];
  const float* key   = (const float*)d_in[1];
  const float* mask  = (const float*)d_in[2];
  const float* Wq    = (const float*)d_in[3];
  const float* bq    = (const float*)d_in[4];
  const float* Wk    = (const float*)d_in[5];
  const float* bk    = (const float*)d_in[6];
  float* out = (float*)d_out;

  _Float16* qh = (_Float16*)d_ws;                       // [B,H,S,DK] f16, 6 MB
  _Float16* kh = qh + (size_t)NB * NH * NS * NDK;       // 6 MB

  // Projections: 256 m-tiles * 24 n-pairs = 6144 waves, 4 waves/block.
  proj_kernel<<<1536, 128, 0, stream>>>(query, Wq, bq, qh);
  proj_kernel<<<1536, 128, 0, stream>>>(key,   Wk, bk, kh);

  // Scores+softmax: B*H*(S/16) = 3072 workgroups of 128 threads.
  attn_kernel<<<3072, 128, 0, stream>>>(qh, kh, mask, out);
}
